// SimpleGATLayer_71975061946651
// MI455X (gfx1250) — compile-verified, matrix operand order checked by reading, not run on hardware
//
#include <hip/hip_runtime.h>
#include <hip/hip_bf16.h>

#define B_DIM   8
#define N_DIM   2048
#define D_DIM   256
#define NEG_INF (-1000000000.0f)
#define LEAKY   0.2f

typedef __attribute__((ext_vector_type(16))) __bf16 v16bf;
typedef __attribute__((ext_vector_type(8)))  float  v8f;

#if __has_builtin(__builtin_amdgcn_global_load_async_to_lds_b128) && \
    __has_builtin(__builtin_amdgcn_s_wait_asynccnt)
#define HAVE_ASYNC_LDS 1
#else
#define HAVE_ASYNC_LDS 0
#endif

#if HAVE_ASYNC_LDS
__device__ __forceinline__ void async_copy_b128(const void* g, void* l) {
    typedef __attribute__((vector_size(16))) int v4i;
    __builtin_amdgcn_global_load_async_to_lds_b128(
        (__attribute__((address_space(1))) v4i*)g,
        (__attribute__((address_space(3))) v4i*)l,
        /*imm offset*/0, /*cpol*/0);
}
#endif

// ---- WMMA fragment loaders (CDNA5 wave32 layouts, 16-bit 16x16x32) -------
// tile stored [row][k]; lanes 0-15: row=lane, K {0..7}+{16..23};
// lanes 16-31: row=lane-16, K {8..15}+{24..31}
__device__ __forceinline__ v16bf frag_rowmajor(const __bf16* t, int row_base,
                                               int stride, int lane) {
    const int r  = row_base + (lane & 15);
    const int kh = (lane >> 4) << 3;
    const __bf16* p = t + r * stride;
    v16bf f;
#pragma unroll
    for (int i = 0; i < 8; ++i) {
        f[i]     = p[kh + i];
        f[i + 8] = p[16 + kh + i];
    }
    return f;
}

// tile stored [k][n]; lane owns column n
__device__ __forceinline__ v16bf frag_colmajor(const __bf16* t, int col_base,
                                               int stride, int lane) {
    const int n  = col_base + (lane & 15);
    const int kh = (lane >> 4) << 3;
    v16bf f;
#pragma unroll
    for (int i = 0; i < 8; ++i) {
        f[i]     = t[(kh + i) * stride + n];
        f[i + 8] = t[(16 + kh + i) * stride + n];
    }
    return f;
}

// ---- Kernel 1: h = x @ W^T  (fp32 in -> bf16 WMMA -> bf16 h) -------------
__global__ void __launch_bounds__(256)
k_gemm_h(const float* __restrict__ x, const float* __restrict__ W,
         __bf16* __restrict__ h) {
    __shared__ __align__(16) __bf16 xs[32 * 40];    // A tile [i][k]
    __shared__ __align__(16) __bf16 wt[256 * 40];   // B tile [o][k]

    const int b    = blockIdx.y;
    const int i0   = blockIdx.x * 32;
    const int tid  = threadIdx.x;
    const int lane = tid & 31;
    const int wave = tid >> 5;
    const int wm   = wave & 1;        // M sub-tile (0..1)
    const int wn   = wave >> 1;       // N strip of 64 cols (0..3)

    const float* xrow = x + (size_t)(b * N_DIM + i0) * D_DIM;

    v8f acc[4] = {};

    for (int k0 = 0; k0 < D_DIM; k0 += 32) {
        {   // x tile 32x32 (1024 f32, 4/thread, coalesced)
            int idx = tid * 4;
            int r = idx >> 5, k = idx & 31;
            const float* src = xrow + r * D_DIM + k0 + k;
#pragma unroll
            for (int j = 0; j < 4; ++j)
                xs[r * 40 + k + j] = (__bf16)src[j];
        }
        // W tile 256x32 (8192 f32, 32/thread, coalesced)
#pragma unroll
        for (int rep = 0; rep < 8; ++rep) {
            int idx = rep * 1024 + tid * 4;
            int o = idx >> 5, k = idx & 31;
            const float* src = W + o * D_DIM + k0 + k;
#pragma unroll
            for (int j = 0; j < 4; ++j)
                wt[o * 40 + k + j] = (__bf16)src[j];
        }
        __syncthreads();

        v16bf afrag = frag_rowmajor(xs, wm * 16, 40, lane);
#pragma unroll
        for (int sub = 0; sub < 4; ++sub) {
            v16bf bfrag = frag_rowmajor(wt, wn * 64 + sub * 16, 40, lane);
            acc[sub] = __builtin_amdgcn_wmma_f32_16x16x32_bf16(
                false, afrag, false, bfrag, (short)0, acc[sub], false, false);
        }
        __syncthreads();
    }

    const int mr = (lane >> 4) << 3;       // C/D: M = v + 8*(lane>=16)
    const int n  = lane & 15;              //      N = lane & 15
#pragma unroll
    for (int sub = 0; sub < 4; ++sub) {
#pragma unroll
        for (int v = 0; v < 8; ++v) {
            int gi = i0 + wm * 16 + mr + v;
            int gd = wn * 64 + sub * 16 + n;
            h[(size_t)(b * N_DIM + gi) * D_DIM + gd] = (__bf16)acc[sub][v];
        }
    }
}

// ---- Kernel 2: el = h@aL, er = h@aR  (wave per row, shuffle reduce) ------
__global__ void __launch_bounds__(256)
k_attn_coef(const __bf16* __restrict__ h, const float* __restrict__ a,
            float* __restrict__ el, float* __restrict__ er) {
    const int lane = threadIdx.x & 31;
    const int wave = threadIdx.x >> 5;
    const int row  = blockIdx.x * 8 + wave;
    const __bf16* hrow = h + (size_t)row * D_DIM;

    float sl = 0.f, sr = 0.f;
#pragma unroll
    for (int k = 0; k < D_DIM; k += 32) {
        float hv = (float)hrow[k + lane];
        sl += hv * a[k + lane];
        sr += hv * a[D_DIM + k + lane];
    }
#pragma unroll
    for (int off = 16; off > 0; off >>= 1) {
        sl += __shfl_down(sl, off, 32);
        sr += __shfl_down(sr, off, 32);
    }
    if (lane == 0) { el[row] = sl; er[row] = sr; }
}

// ---- Kernel 3: online softmax stats (row max m, row sum s) ---------------
__global__ void __launch_bounds__(256)
k_softmax_stats(const int* __restrict__ adj, const float* __restrict__ el,
                const float* __restrict__ er, float* __restrict__ mrow,
                float* __restrict__ srow) {
    const int lane = threadIdx.x & 31;
    const int wave = threadIdx.x >> 5;
    const int row  = blockIdx.x * 8 + wave;     // b*N + i
    const int b    = row >> 11;

    const float e_i = el[row];
    const int*   arow = adj + (size_t)row * N_DIM;
    const float* erb  = er + b * N_DIM;

    float m = NEG_INF, s = 0.f;
    for (int j = lane; j < N_DIM; j += 32) {
        float e = e_i + erb[j];
        e = e > 0.f ? e : LEAKY * e;
        e = arow[j] ? e : NEG_INF;
        float mn = fmaxf(m, e);
        s = s * __expf(m - mn) + __expf(e - mn);
        m = mn;
    }
#pragma unroll
    for (int off = 16; off > 0; off >>= 1) {
        float m2 = __shfl_down(m, off, 32);
        float s2 = __shfl_down(s, off, 32);
        float mn = fmaxf(m, m2);
        s = s * __expf(m - mn) + s2 * __expf(m2 - mn);
        m = mn;
    }
    if (lane == 0) { mrow[row] = m; srow[row] = s; }
}

// ---- Kernel 4: out = alpha @ h -------------------------------------------
// 64(i) x 256(d) block tile, K = j in steps of 32.  8 waves = 4(M) x 2(N),
// 8 WMMA subtiles/wave.  h tiles double-buffered in LDS and fetched with
// GLOBAL_LOAD_ASYNC_TO_LDS_B128 (ASYNCcnt) one K-step ahead of the WMMAs.
__global__ void __launch_bounds__(256)
k_attn_out(const int* __restrict__ adj, const __bf16* __restrict__ h,
           const float* __restrict__ el, const float* __restrict__ er,
           const float* __restrict__ mrow, const float* __restrict__ srow,
           float* __restrict__ out) {
    __shared__ __align__(16) __bf16 at[64 * 40];        // alpha tile [i][j]
    __shared__ __align__(16) __bf16 bt[2][32 * 264];    // h tile [j][d], x2

    const int b    = blockIdx.y;
    const int i0   = blockIdx.x * 64;
    const int tid  = threadIdx.x;
    const int lane = tid & 31;
    const int wave = tid >> 5;
    const int wm   = wave >> 1;       // 0..3 -> M sub-tile
    const int wn   = wave & 1;        // 0..1 -> N strip of 128 cols

    // each thread owns one alpha row (r = tid/4) and 8 consecutive j's
    const int ar = tid >> 2;          // 0..63
    const int ac = (tid & 3) * 8;     // 0,8,16,24
    const int grow  = b * N_DIM + i0 + ar;
    const float e_i   = el[grow];
    const float m_i   = mrow[grow];
    const float inv_s = 1.f / srow[grow];
    const int*    arow = adj + (size_t)grow * N_DIM;
    const float*  erb  = er + b * N_DIM;
    const __bf16* hb   = h + (size_t)b * N_DIM * D_DIM;

    // per-thread h-tile chunk coordinates (4 x 16B chunks per tile)
    const int hj[4] = { (tid * 8) >> 8,         (2048 + tid * 8) >> 8,
                        (4096 + tid * 8) >> 8,  (6144 + tid * 8) >> 8 };
    const int hd = (tid * 8) & 255;

    v8f acc[8] = {};

    // prologue: stage h tile for j0 = 0 into buffer 0
#pragma unroll
    for (int rep = 0; rep < 4; ++rep) {
        const __bf16* src = hb + (size_t)hj[rep] * D_DIM + hd;
        __bf16* dst = &bt[0][hj[rep] * 264 + hd];
#if HAVE_ASYNC_LDS
        async_copy_b128(src, dst);
#else
#pragma unroll
        for (int q = 0; q < 8; ++q) dst[q] = src[q];
#endif
    }

    for (int t = 0; t < N_DIM / 32; ++t) {
        const int j0  = t * 32;
        const int buf = t & 1;

        // stage next h tile into the other buffer (overlaps with WMMA below)
        if (t + 1 < N_DIM / 32) {
#pragma unroll
            for (int rep = 0; rep < 4; ++rep) {
                const __bf16* src = hb + (size_t)(j0 + 32 + hj[rep]) * D_DIM + hd;
                __bf16* dst = &bt[buf ^ 1][hj[rep] * 264 + hd];
#if HAVE_ASYNC_LDS
                async_copy_b128(src, dst);
#else
#pragma unroll
                for (int q = 0; q < 8; ++q) dst[q] = src[q];
#endif
            }
            __builtin_prefetch(&arow[j0 + 32 + ac], 0, 0);  // next adj tile
        }

        // alpha tile: exp(masked_e - m)/s  (exact for all-masked rows)
#pragma unroll
        for (int j = 0; j < 8; ++j) {
            int gj = j0 + ac + j;
            float e = e_i + erb[gj];
            e = e > 0.f ? e : LEAKY * e;
            e = arow[gj] ? e : NEG_INF;
            at[ar * 40 + ac + j] = (__bf16)(__expf(e - m_i) * inv_s);
        }

#if HAVE_ASYNC_LDS
        // current tile's 4 loads done; next tile's 4 may stay in flight
        if (t + 1 < N_DIM / 32) __builtin_amdgcn_s_wait_asynccnt(4);
        else                    __builtin_amdgcn_s_wait_asynccnt(0);
#endif
        __syncthreads();

        v16bf afrag = frag_rowmajor(at, wm * 16, 40, lane);
#pragma unroll
        for (int sub = 0; sub < 8; ++sub) {
            v16bf bfrag = frag_colmajor(bt[buf], wn * 128 + sub * 16, 264, lane);
            acc[sub] = __builtin_amdgcn_wmma_f32_16x16x32_bf16(
                false, afrag, false, bfrag, (short)0, acc[sub], false, false);
        }
        __syncthreads();
    }

    const int mr = (lane >> 4) << 3;
    const int n  = lane & 15;
#pragma unroll
    for (int sub = 0; sub < 8; ++sub) {
#pragma unroll
        for (int v = 0; v < 8; ++v) {
            int gi = i0 + wm * 16 + mr + v;
            int gd = wn * 128 + sub * 16 + n;
            out[(size_t)(b * N_DIM + gi) * D_DIM + gd] = acc[sub][v];
        }
    }
}

extern "C" void kernel_launch(void* const* d_in, const int* in_sizes, int n_in,
                              void* d_out, int out_size, void* d_ws, size_t ws_size,
                              hipStream_t stream) {
    (void)in_sizes; (void)n_in; (void)out_size; (void)ws_size;

    const float* x   = (const float*)d_in[0];
    const int*   adj = (const int*)d_in[1];
    const float* W   = (const float*)d_in[2];
    const float* a   = (const float*)d_in[3];
    float*       out = (float*)d_out;

    char* p = (char*)d_ws;
    __bf16* h  = (__bf16*)p; p += (size_t)B_DIM * N_DIM * D_DIM * sizeof(__bf16);
    float*  el = (float*)p;  p += (size_t)B_DIM * N_DIM * sizeof(float);
    float*  er = (float*)p;  p += (size_t)B_DIM * N_DIM * sizeof(float);
    float*  mr = (float*)p;  p += (size_t)B_DIM * N_DIM * sizeof(float);
    float*  sr = (float*)p;  p += (size_t)B_DIM * N_DIM * sizeof(float);

    k_gemm_h       <<<dim3(N_DIM / 32, B_DIM), 256, 0, stream>>>(x, W, h);
    k_attn_coef    <<<dim3(B_DIM * N_DIM / 8), 256, 0, stream>>>(h, a, el, er);
    k_softmax_stats<<<dim3(B_DIM * N_DIM / 8), 256, 0, stream>>>(adj, el, er, mr, sr);
    k_attn_out     <<<dim3(N_DIM / 64, B_DIM), 256, 0, stream>>>(adj, h, el, er, mr, sr, out);
}
